// FloorplanGNN_5789615915295
// MI455X (gfx1250) — matrix operand order repaired and checked by reading, not compiled, not run on hardware
//
#include <hip/hip_runtime.h>
#include <hip/hip_bf16.h>

typedef _Float16 v16h __attribute__((ext_vector_type(16)));
typedef _Float16 h8   __attribute__((ext_vector_type(8)));
typedef float    v8f  __attribute__((ext_vector_type(8)));

// ---------------- utility ----------------
__device__ __forceinline__ unsigned fkey(float f) {
    unsigned u = __float_as_uint(f);
    return (u & 0x80000000u) ? ~u : (u | 0x80000000u);
}
__device__ __forceinline__ float keyf(unsigned u) {
    return __uint_as_float((u & 0x80000000u) ? (u & 0x7fffffffu) : ~u);
}

__global__ void k_zero(float* p, long long n) {
    long long i = (long long)blockIdx.x * blockDim.x + threadIdx.x;
    long long s = (long long)gridDim.x * blockDim.x;
    for (; i < n; i += s) p[i] = 0.0f;
}

// sum of edge_attr columns (E x 2) -> ea_sum[2]
__global__ void k_easum(const float* __restrict__ ea, long long E, float* ea_sum) {
    __shared__ float sh[512];
    int t = threadIdx.x;
    float a0 = 0.f, a1 = 0.f;
    long long i = (long long)blockIdx.x * blockDim.x + t;
    long long s = (long long)gridDim.x * blockDim.x;
    for (; i < E; i += s) { a0 += ea[2 * i]; a1 += ea[2 * i + 1]; }
    sh[t] = a0; sh[256 + t] = a1;
    __syncthreads();
    for (int d = 128; d >= 1; d >>= 1) {
        if (t < d) { sh[t] += sh[t + d]; sh[256 + t] += sh[256 + t + d]; }
        __syncthreads();
    }
    if (t == 0) { atomicAdd(&ea_sum[0], sh[0]); atomicAdd(&ea_sum[1], sh[256]); }
}

// precompute C1[2][4], sl1[4], C2[2][2], sl2[2] into sm[]
// layout: C1 at 0 (f*4+h), sl1 at 8, C2 at 12 (f*2+h), sl2 at 16
__global__ void k_prep(const float* We1, const float* ae1,
                       const float* We2, const float* ae2,
                       const float* ea_sum, float invE, float* sm) {
    int t = threadIdx.x;
    if (t < 8) {
        int f = t >> 2, h = t & 3; float s = 0.f;
        for (int d = 0; d < 64; d++) s += We1[f * 256 + h * 64 + d] * ae1[h * 64 + d];
        sm[t] = s;
    } else if (t < 12) {
        int u = t - 8; int f = u >> 1, h = u & 1; float s = 0.f;
        for (int d = 0; d < 64; d++) s += We2[f * 128 + h * 64 + d] * ae2[h * 64 + d];
        sm[12 + u] = s;
    }
    __syncthreads();
    float m0 = ea_sum[0] * invE, m1 = ea_sum[1] * invE;
    if (t < 4)      sm[8 + t]        = m0 * sm[t]            + m1 * sm[4 + t];
    else if (t < 6) sm[16 + (t - 4)] = m0 * sm[12 + (t - 4)] + m1 * sm[14 + (t - 4)];
}

// layer1 linear: h1[n,c] = sum_f x[n,f] W1[f,c]   (K=3, bandwidth bound)
__global__ void k_lin1(const float* __restrict__ x, const float* __restrict__ W1,
                       float* __restrict__ h1, long long N) {
    long long T = N * 256;
    long long i = (long long)blockIdx.x * blockDim.x + threadIdx.x;
    long long s = (long long)gridDim.x * blockDim.x;
    for (; i < T; i += s) {
        long long n = i >> 8; int c = (int)(i & 255);
        h1[i] = x[n * 3] * W1[c] + x[n * 3 + 1] * W1[256 + c] + x[n * 3 + 2] * W1[512 + c];
    }
}

// per-node attention dots: als[n,h]=feat[n,h,:].a_src[h], ald likewise. blockDim=H*64
__global__ void k_attdot(const float* __restrict__ feat, const float* __restrict__ as_,
                         const float* __restrict__ ad_, float* als, float* ald, int H) {
    __shared__ float ss[256], sd[256];
    long long n = blockIdx.x;
    int t = threadIdx.x, C = blockDim.x;
    float f = feat[n * C + t];
    ss[t] = f * as_[t]; sd[t] = f * ad_[t];
    __syncthreads();
    for (int d = 32; d >= 1; d >>= 1) {
        if ((t & 63) < d) { ss[t] += ss[t + d]; sd[t] += sd[t + d]; }
        __syncthreads();
    }
    if ((t & 63) == 0) { als[n * H + (t >> 6)] = ss[t]; ald[n * H + (t >> 6)] = sd[t]; }
}

// edge alpha + segment max (float via monotone uint key)
__global__ void k_alpha(const int* __restrict__ ei, const float* __restrict__ ea,
                        const float* __restrict__ als, const float* __restrict__ ald,
                        const float* __restrict__ C, const float* __restrict__ sl,
                        float* abuf, unsigned* mkey, int E, int N, int H) {
    long long T = (long long)(E + N) * H;
    long long i = (long long)blockIdx.x * blockDim.x + threadIdx.x;
    long long s = (long long)gridDim.x * blockDim.x;
    for (; i < T; i += s) {
        int h = (int)(i % H); long long e = i / H;
        int src, dst; float ale;
        if (e < E) {
            src = ei[e]; dst = ei[E + e];
            ale = ea[e * 2] * C[h] + ea[e * 2 + 1] * C[H + h];
        } else {
            src = dst = (int)(e - E); ale = sl[h];
        }
        float a = als[(long long)src * H + h] + ald[(long long)dst * H + h] + ale;
        a = a > 0.f ? a : 0.2f * a;           // leaky relu
        abuf[e * H + h] = a;
        atomicMax(&mkey[(long long)dst * H + h], fkey(a));
    }
}

// exp(alpha - max) + segment sum
__global__ void k_expsum(const int* __restrict__ ei, float* abuf,
                         const unsigned* __restrict__ mkey, float* den,
                         int E, int N, int H) {
    long long T = (long long)(E + N) * H;
    long long i = (long long)blockIdx.x * blockDim.x + threadIdx.x;
    long long s = (long long)gridDim.x * blockDim.x;
    for (; i < T; i += s) {
        int h = (int)(i % H); long long e = i / H;
        int dst = (e < E) ? ei[E + e] : (int)(e - E);
        float ex = __expf(abuf[i] - keyf(mkey[(long long)dst * H + h]));
        abuf[i] = ex;
        atomicAdd(&den[(long long)dst * H + h], ex);
    }
}

// message scatter: agg[dst,:] += (ex/den) * feat[src,:]. blockDim = H*64, grid = E+N
// agg fits in the 192MB L2 -> atomics resolve in L2.
__global__ void k_scatter(const int* __restrict__ ei, const float* __restrict__ feat,
                          const float* __restrict__ abuf, const float* __restrict__ den,
                          float* agg, int E, int N, int H) {
    long long e = blockIdx.x;
    int t = threadIdx.x, C = blockDim.x;
    int src, dst;
    if (e < E) { src = ei[e]; dst = ei[E + e]; } else { src = dst = (int)(e - E); }
    int h = t >> 6;
    float w = abuf[e * H + h] / den[(long long)dst * H + h];
    atomicAdd(&agg[(long long)dst * C + t], w * feat[(long long)src * C + t]);
}

// column sums / sumsq of y = agg + bias (for GraphNorm). blockDim = C
__global__ void k_colstats(const float* __restrict__ agg, const float* __restrict__ bias,
                           float* sums, int N, int C, int rpb) {
    int c = threadIdx.x;
    int r0 = blockIdx.x * rpb;
    int r1 = r0 + rpb; if (r1 > N) r1 = N;
    float s = 0.f, q = 0.f;
    for (int r = r0; r < r1; r++) {
        float y = agg[(long long)r * C + c] + bias[c];
        s += y; q += y * y;
    }
    atomicAdd(&sums[c], s);
    atomicAdd(&sums[C + c], q);
}

// GraphNorm + ELU, emit f16 for the WMMA GEMM that follows
__global__ void k_norm_elu(const float* __restrict__ agg, const float* __restrict__ bias,
                           const float* __restrict__ sums, const float* __restrict__ gw,
                           const float* __restrict__ gb, const float* __restrict__ gms,
                           _Float16* z, long long N, int C, float invN) {
    long long T = N * C;
    long long i = (long long)blockIdx.x * blockDim.x + threadIdx.x;
    long long st = (long long)gridDim.x * blockDim.x;
    for (; i < T; i += st) {
        int c = (int)(i % C);
        float mu = sums[c] * invN, ey2 = sums[C + c] * invN;
        float mm = mu * gms[c];
        float var = ey2 - 2.f * mm * mu + mm * mm;
        float y = agg[i] + bias[c];
        float o = (y - mm) * __frsqrt_rn(var + 1e-5f) * gw[c] + gb[c];
        o = o > 0.f ? o : __expf(o) - 1.f;    // ELU
        z[i] = (_Float16)o;
    }
}

// Repack B (row-major K x srcCols f32, leading dim srcLd) into the per-lane WMMA
// fragment layout, f16, padded to Nc columns (multiple of 16):
//   Bp[ ((ks*numCT + ct)*32 + lane)*16 + j ] = B[(ks*32 + (lane>=16)*8 + (j&7) + (j>>3)*16)*srcLd + ct*16 + (lane&15)]
// so the GEMM inner loop reads two contiguous 16B vectors per lane per k-step.
__global__ void k_brepack(const float* __restrict__ B, _Float16* __restrict__ Bp,
                          int K, int Nc, int srcLd, int srcCols) {
    int numCT = Nc >> 4;
    int total = K * Nc;
    int i = blockIdx.x * blockDim.x + threadIdx.x;
    if (i >= total) return;
    int j = i & 15;
    int l = (i >> 4) & 31;
    int rest = i >> 9;
    int ct = rest % numCT;
    int ks = rest / numCT;
    int krow = ks * 32 + ((l >> 4) * 8) + (j & 7) + ((j >> 3) * 16);
    int col = ct * 16 + (l & 15);
    float v = (col < srcCols && krow < K) ? B[krow * srcLd + col] : 0.f;
    Bp[i] = (_Float16)v;
}

// WMMA f16 GEMM: C[M,Nc] = A[M,K] * Bp (+bias). One wave per 16x16 tile,
// blockDim = 32*(Nc/16). K multiple of 32. A per CDNA5 16-bit A-frag layout:
// lane<16 holds K {0-7,16-23}, lane>=16 holds K {8-15,24-31} -> two contiguous
// 16B loads per lane. B comes pre-repacked (k_brepack) -> two contiguous 16B loads.
__global__ void k_gemm_wmma(const _Float16* __restrict__ A, const _Float16* __restrict__ Bp,
                            float* __restrict__ Cout, int Nrows, int K, int Nc,
                            int ldc, int colLimit, const float* __restrict__ bias) {
    int lane = threadIdx.x & 31;
    int wv = threadIdx.x >> 5;
    int numCT = Nc >> 4;
    int row0 = blockIdx.x * 16;

    int arow = row0 + (lane & 15);
    if (arow >= Nrows) arow = Nrows - 1;          // clamp loads at M edge
    int kb = (lane >> 4) * 8;                     // A sub-K offset per lane half
    const _Float16* ab = A + (long long)arow * K + kb;
    const _Float16* bb = Bp + ((size_t)wv * 32 + lane) * 16;
    size_t bstride = (size_t)numCT * 32 * 16;     // halves per k-step

    v8f acc = {};
    int nks = K >> 5;
#pragma unroll 4
    for (int ks = 0; ks < nks; ks++) {
        h8 alo = *(const h8*)(ab + (ks << 5));
        h8 ahi = *(const h8*)(ab + (ks << 5) + 16);
        v16h a = __builtin_shufflevector(alo, ahi, 0, 1, 2, 3, 4, 5, 6, 7,
                                         8, 9, 10, 11, 12, 13, 14, 15);
        const h8* bp = (const h8*)(bb + ks * bstride);
        h8 blo = bp[0];
        h8 bhi = bp[1];
        v16h b = __builtin_shufflevector(blo, bhi, 0, 1, 2, 3, 4, 5, 6, 7,
                                         8, 9, 10, 11, 12, 13, 14, 15);
        acc = __builtin_amdgcn_wmma_f32_16x16x32_f16(false, a, false, b,
                                                     (short)0, acc, false, false);
    }
    union { v8f v; float e[8]; } cu; cu.v = acc;
    int col = wv * 16 + (lane & 15);
    int radd = (lane >> 4) * 8;
    if (col < colLimit) {
        float bb2 = bias ? bias[col] : 0.f;
#pragma unroll
        for (int r = 0; r < 8; r++) {
            int row = row0 + r + radd;
            if (row < Nrows) Cout[(long long)row * ldc + col] = cu.e[r] + bb2;
        }
    }
}

// ---------------- host launch ----------------
extern "C" void kernel_launch(void* const* d_in, const int* in_sizes, int n_in,
                              void* d_out, int out_size, void* d_ws, size_t ws_size,
                              hipStream_t stream) {
    const float* x    = (const float*)d_in[0];
    const int*   ei   = (const int*)d_in[1];
    const float* ea   = (const float*)d_in[2];
    const float* W1   = (const float*)d_in[3];
    const float* We1  = (const float*)d_in[4];
    const float* as1  = (const float*)d_in[5];
    const float* ad1  = (const float*)d_in[6];
    const float* ae1  = (const float*)d_in[7];
    const float* b1   = (const float*)d_in[8];
    const float* g1w  = (const float*)d_in[9];
    const float* g1b  = (const float*)d_in[10];
    const float* g1ms = (const float*)d_in[11];
    const float* W2   = (const float*)d_in[12];
    const float* We2  = (const float*)d_in[13];
    const float* as2  = (const float*)d_in[14];
    const float* ad2  = (const float*)d_in[15];
    const float* ae2  = (const float*)d_in[16];
    const float* b2   = (const float*)d_in[17];
    const float* g2w  = (const float*)d_in[18];
    const float* g2b  = (const float*)d_in[19];
    const float* g2ms = (const float*)d_in[20];
    const float* Wc   = (const float*)d_in[21];
    const float* bc   = (const float*)d_in[22];

    const long long N = in_sizes[0] / 3;
    const int E = in_sizes[1] / 2;
    const long long T = E + N;

    float* w = (float*)d_ws;
    size_t o = 0;
    auto alloc = [&](size_t nf) { size_t r = o; o += (nf + 7) & ~(size_t)7; return r; };
    // zero region (accumulators)
    size_t ea_sum = alloc(4);
    size_t mkey1  = alloc(N * 4);
    size_t den1   = alloc(N * 4);
    size_t agg1   = alloc(N * 256);
    size_t sums1  = alloc(512);
    size_t mkey2  = alloc(N * 2);
    size_t den2   = alloc(N * 2);
    size_t agg2   = alloc(N * 128);
    size_t sums2  = alloc(256);
    size_t zend = o;
    // scratch
    size_t h1   = alloc(N * 256);
    size_t als1 = alloc(N * 4);
    size_t ald1 = alloc(N * 4);
    size_t ab1  = alloc(T * 4);
    size_t sm   = alloc(32);
    size_t z1   = alloc(N * 128);   // f16, N*256 halves
    size_t w2p  = alloc(256 * 128 / 2);  // repacked W2, f16
    size_t h2   = alloc(N * 128);
    size_t als2 = alloc(N * 2);
    size_t ald2 = alloc(N * 2);
    size_t ab2  = alloc(T * 2);
    size_t z2   = alloc(N * 64);    // f16, N*128 halves
    size_t wcp  = alloc(128 * 16 / 2);   // repacked padded Wc, f16
    (void)ws_size; (void)n_in; (void)out_size;

    const int GS = 2048;
    // init accumulators
    k_zero<<<GS, 256, 0, stream>>>(w, (long long)zend);
    // edge-attr mean + tiny attention constants
    k_easum<<<512, 256, 0, stream>>>(ea, E, w + ea_sum);
    k_prep<<<1, 32, 0, stream>>>(We1, ae1, We2, ae2, w + ea_sum, 1.0f / E, w + sm);
    // ---- layer 1 ----
    k_lin1<<<GS, 256, 0, stream>>>(x, W1, w + h1, N);
    k_attdot<<<(int)N, 256, 0, stream>>>(w + h1, as1, ad1, w + als1, w + ald1, 4);
    k_alpha<<<GS, 256, 0, stream>>>(ei, ea, w + als1, w + ald1, w + sm, w + sm + 8,
                                    w + ab1, (unsigned*)(w + mkey1), E, (int)N, 4);
    k_expsum<<<GS, 256, 0, stream>>>(ei, w + ab1, (unsigned*)(w + mkey1), w + den1, E, (int)N, 4);
    k_scatter<<<(int)T, 256, 0, stream>>>(ei, w + h1, w + ab1, w + den1, w + agg1, E, (int)N, 4);
    k_colstats<<<(int)((N + 127) / 128), 256, 0, stream>>>(w + agg1, b1, w + sums1, (int)N, 256, 128);
    k_norm_elu<<<GS, 256, 0, stream>>>(w + agg1, b1, w + sums1, g1w, g1b, g1ms,
                                       (_Float16*)(w + z1), N, 256, 1.0f / (float)N);
    // ---- GEMM: z1[N,256] @ W2[256,128] via v_wmma_f32_16x16x32_f16 ----
    k_brepack<<<(256 * 128 + 255) / 256, 256, 0, stream>>>(W2, (_Float16*)(w + w2p),
                                                           256, 128, 128, 128);
    k_gemm_wmma<<<(int)((N + 15) / 16), 256, 0, stream>>>(
        (const _Float16*)(w + z1), (const _Float16*)(w + w2p), w + h2,
        (int)N, 256, 128, 128, 128, nullptr);
    // ---- layer 2 ----
    k_attdot<<<(int)N, 128, 0, stream>>>(w + h2, as2, ad2, w + als2, w + ald2, 2);
    k_alpha<<<GS, 256, 0, stream>>>(ei, ea, w + als2, w + ald2, w + sm + 12, w + sm + 16,
                                    w + ab2, (unsigned*)(w + mkey2), E, (int)N, 2);
    k_expsum<<<GS, 256, 0, stream>>>(ei, w + ab2, (unsigned*)(w + mkey2), w + den2, E, (int)N, 2);
    k_scatter<<<(int)T, 128, 0, stream>>>(ei, w + h2, w + ab2, w + den2, w + agg2, E, (int)N, 2);
    k_colstats<<<(int)((N + 127) / 128), 128, 0, stream>>>(w + agg2, b2, w + sums2, (int)N, 128, 128);
    k_norm_elu<<<GS, 256, 0, stream>>>(w + agg2, b2, w + sums2, g2w, g2b, g2ms,
                                       (_Float16*)(w + z2), N, 128, 1.0f / (float)N);
    // ---- classifier: z2[N,128] @ Wc[128,13] (padded to 16) via WMMA ----
    k_brepack<<<(128 * 16 + 255) / 256, 256, 0, stream>>>(Wc, (_Float16*)(w + wcp),
                                                          128, 16, 13, 13);
    k_gemm_wmma<<<(int)((N + 15) / 16), 32, 0, stream>>>(
        (const _Float16*)(w + z2), (const _Float16*)(w + wcp), (float*)d_out,
        (int)N, 128, 16, 13, 13, bc);
}
